// FLK_NKRR_61392262529364
// MI455X (gfx1250) — matrix operand order repaired and checked by reading, not compiled, not run on hardware
//
#include <hip/hip_runtime.h>
#include <hip/hip_bf16.h>

// ---------------------------------------------------------------------------
// Gaussian RBF kernel mat-vec forward (Falkon/NKRR style), CDNA5 / gfx1250.
//   preds = exp(-(|x|^2 + |c|^2 - 2 X C^T) / (2 sigma^2)) @ alpha
//   loss  = mean((preds - Y)^2)
// Strategy:
//   * f32 -> bf16 hi/lo split; each 16x16x32 K-slab = 3x
//     v_wmma_f32_16x16x32_bf16 (hi*hi + hi*lo + lo*hi) ~ fp32 accuracy at
//     bf16 matrix-core rates.
//   * A fragments (16 rows x 256 K, hi+lo) live entirely in VGPRs for the
//     whole M sweep.
//   * B tile (shared by all 8 waves) is staged into LDS once per block via
//     async DMA (global_load_async_to_lds_b128, ASYNCcnt), double-buffered
//     so the DMA of tile mt+1 overlaps the WMMAs of tile mt; inner loop
//     reads fragments with ds_load_b128 (short latency) instead of L2.
// ---------------------------------------------------------------------------

typedef __bf16 v16bf __attribute__((ext_vector_type(16)));
typedef float  v8f   __attribute__((ext_vector_type(8)));
typedef int    v4i_  __attribute__((__vector_size__(16)));
typedef __attribute__((address_space(1))) v4i_* as1_v4i_p;   // global
typedef __attribute__((address_space(3))) v4i_* as3_v4i_p;   // LDS

#define DCONST        256     // feature dim (reference: D = 256)
#define ROWS_PER_BLK  128     // X rows per workgroup (8 waves x 16 rows)
#define MAIN_THREADS  256
#define KCHUNKS       (DCONST / 32)   // 8
#define TILE_ELEMS    (16 * DCONST)   // one B tile (16 centers x 256), 4096

// ---- async global->LDS copy of 16 bytes (ASYNCcnt-tracked) ----------------
__device__ __forceinline__ void async_b128(const void* g, void* l) {
#if __has_builtin(__builtin_amdgcn_global_load_async_to_lds_b128)
    __builtin_amdgcn_global_load_async_to_lds_b128(
        (as1_v4i_p)(unsigned long long)g,
        (as3_v4i_p)(unsigned int)(unsigned long long)l,
        0, 0);
#else
    const unsigned int laddr = (unsigned int)(unsigned long long)l;
    asm volatile("global_load_async_to_lds_b128 %0, %1, off"
                 :: "v"(laddr), "v"(g) : "memory");
#endif
}

__device__ __forceinline__ void wait_async0() {
#if __has_builtin(__builtin_amdgcn_s_wait_asynccnt)
    __builtin_amdgcn_s_wait_asynccnt(0);
#else
    asm volatile("s_wait_asynccnt 0x0" ::: "memory");
#endif
}

// ---------------------------------------------------------------------------
// Kernel 1: split centers into bf16 hi/lo and compute |c|^2 per center.
// One block per center row, 256 threads == D elements.
// ---------------------------------------------------------------------------
__global__ void flk_prep_centers(const float* __restrict__ centers,
                                 __bf16* __restrict__ Chi,
                                 __bf16* __restrict__ Clo,
                                 float* __restrict__ c2) {
    const int m   = blockIdx.x;
    const int t   = threadIdx.x;
    const float x = centers[(size_t)m * DCONST + t];

    const __bf16 hi = (__bf16)x;
    const float  hf = (float)hi;
    const __bf16 lo = (__bf16)(x - hf);
    Chi[(size_t)m * DCONST + t] = hi;
    Clo[(size_t)m * DCONST + t] = lo;

    float s = x * x;
    s += __shfl_xor(s, 16, 32);
    s += __shfl_xor(s,  8, 32);
    s += __shfl_xor(s,  4, 32);
    s += __shfl_xor(s,  2, 32);
    s += __shfl_xor(s,  1, 32);
    __shared__ float red[8];
    if ((t & 31) == 0) red[t >> 5] = s;
    __syncthreads();
    if (t == 0) {
        float tot = 0.f;
        #pragma unroll
        for (int w = 0; w < 8; ++w) tot += red[w];
        c2[m] = tot;
    }
}

// ---------------------------------------------------------------------------
// Kernel 2: main fused RBF mat-vec.
//   grid.x = ceil(N/128), block = 256 (8 wave32).
//   Wave w owns rows [blk*128 + w*16, +16); lane l (and l+16) own row (l&15).
// ---------------------------------------------------------------------------
__global__ void __launch_bounds__(MAIN_THREADS)
flk_rbf_mv(const float* __restrict__ X,
           const float* __restrict__ sigma_p,
           const __bf16* __restrict__ Chi,
           const __bf16* __restrict__ Clo,
           const float* __restrict__ c2,
           const float* __restrict__ alpha,
           float* __restrict__ preds,
           int N, int M) {
    // double-buffered B tile: [buf][hi/lo][16 x 256] bf16 = 32 KB total
    __shared__ __bf16 ldsB[2][2][TILE_ELEMS];

    const int tid  = threadIdx.x;
    const int blk0 = blockIdx.x * ROWS_PER_BLK;
    const int wave = tid >> 5;
    const int lane = tid & 31;
    const int half = lane >> 4;   // 0: K {0-15} of each 32-chunk, 1: {16-31}
    const int l16  = lane & 15;

    const float sg     = sigma_p[0];
    const float inv2s2 = 1.0f / (2.0f * sg * sg);

    // ---- load this lane's A row slices straight from global X, convert to
    //      bf16 hi/lo in registers, accumulate partial |x|^2 on the fly ----
    const int   gr_a = blk0 + wave * 16 + l16;      // row this lane feeds to A
    const float* xrow = X + (size_t)gr_a * DCONST;
    const bool  arow_ok = (gr_a < N);

    v16bf Ahi[KCHUNKS], Alo[KCHUNKS];
    float x2part = 0.f;
    #pragma unroll
    for (int kc = 0; kc < KCHUNKS; ++kc) {
        // A k order used here (B matches): lanes 0-15 hold K=[k0,k0+16),
        // lanes 16-31 hold K=[k0+16,k0+32).
        const int o0 = kc * 32 + half * 16;
        float f[16];
        #pragma unroll
        for (int q4 = 0; q4 < 2; ++q4) {            // two float4 per phase
            float4 a = arow_ok ? *(const float4*)(xrow + o0 + q4 * 4)
                               : make_float4(0.f, 0.f, 0.f, 0.f);
            float4 b = arow_ok ? *(const float4*)(xrow + o0 + 8 + q4 * 4)
                               : make_float4(0.f, 0.f, 0.f, 0.f);
            f[q4 * 4 + 0] = a.x; f[q4 * 4 + 1] = a.y;
            f[q4 * 4 + 2] = a.z; f[q4 * 4 + 3] = a.w;
            f[8 + q4 * 4 + 0] = b.x; f[8 + q4 * 4 + 1] = b.y;
            f[8 + q4 * 4 + 2] = b.z; f[8 + q4 * 4 + 3] = b.w;
        }
        v16bf hi, lo;
        #pragma unroll
        for (int q = 0; q < 16; ++q) {
            const float  x = f[q];
            x2part += x * x;
            const __bf16 h = (__bf16)x;
            hi[q] = h;
            lo[q] = (__bf16)(x - (float)h);
        }
        Ahi[kc] = hi;
        Alo[kc] = lo;
    }
    // row r's |x|^2 is split across lanes r and r+16 (K halves): combine,
    // then fetch the 8 row norms this lane's accumulator VGPRs represent.
    x2part += __shfl_xor(x2part, 16, 32);
    float x2r[8];
    #pragma unroll
    for (int v = 0; v < 8; ++v)
        x2r[v] = __shfl(x2part, half * 8 + v, 32);

    // ---- async staging helpers: 16 KB tile, 256 threads x 64 B each -------
    // threads 0-127 move the hi half (8 KB), threads 128-255 the lo half.
    const int  t128   = tid & 127;
    const bool is_lo  = (tid >= 128);

    auto issue_tile = [&](int mt, int buf) {
        const __bf16* src = (is_lo ? Clo : Chi) + (size_t)mt * TILE_ELEMS
                            + t128 * 32;                    // 32 elems = 64 B
        __bf16* dst = &ldsB[buf][is_lo ? 1 : 0][t128 * 32];
        #pragma unroll
        for (int j = 0; j < 4; ++j)                         // 4 x 16 B
            async_b128(src + j * 8, dst + j * 8);
    };

    // ---- sweep all M in 16-column tiles, double-buffered ----
    v8f partial = {0.f, 0.f, 0.f, 0.f, 0.f, 0.f, 0.f, 0.f};
    const int ntiles = M >> 4;

    issue_tile(0, 0);
    wait_async0();
    __syncthreads();

    for (int mt = 0; mt < ntiles; ++mt) {
        const int buf = mt & 1;
        if (mt + 1 < ntiles)
            issue_tile(mt + 1, buf ^ 1);   // DMA next tile during compute

        // B fragment from LDS: lane = column, lanes 16-31 take K+16.
        const __bf16* bhp = &ldsB[buf][0][l16 * DCONST + (half << 4)];
        const __bf16* blp = &ldsB[buf][1][l16 * DCONST + (half << 4)];

        v8f acc = {0.f, 0.f, 0.f, 0.f, 0.f, 0.f, 0.f, 0.f};
        #pragma unroll
        for (int kc = 0; kc < KCHUNKS; ++kc) {
            const v16bf bh = *(const v16bf*)(bhp + kc * 32);
            const v16bf bl = *(const v16bf*)(blp + kc * 32);
            // fp32-fidelity product: hi*hi + hi*lo + lo*hi
            acc = __builtin_amdgcn_wmma_f32_16x16x32_bf16(
                false, Ahi[kc], false, bh, (short)0, acc, false, false);
            acc = __builtin_amdgcn_wmma_f32_16x16x32_bf16(
                false, Ahi[kc], false, bl, (short)0, acc, false, false);
            acc = __builtin_amdgcn_wmma_f32_16x16x32_bf16(
                false, Alo[kc], false, bh, (short)0, acc, false, false);
        }

        // epilogue: k = exp(-(x2 + c2 - 2 acc)/(2 s^2)); partial += k*alpha
        const float c2c = c2[mt * 16 + l16];
        const float al  = alpha[mt * 16 + l16];
        #pragma unroll
        for (int v = 0; v < 8; ++v) {
            const float t = (2.0f * acc[v] - x2r[v] - c2c) * inv2s2;
            partial[v] += __expf(t) * al;
        }

        wait_async0();      // next tile's DMA has landed
        __syncthreads();    // all waves done reading buf before it's reused
    }

    // reduce each row's partial across the 16 lanes of its half, store preds
    #pragma unroll
    for (int v = 0; v < 8; ++v) {
        float p = partial[v];
        p += __shfl_xor(p, 1, 32);
        p += __shfl_xor(p, 2, 32);
        p += __shfl_xor(p, 4, 32);
        p += __shfl_xor(p, 8, 32);
        const int gr = blk0 + wave * 16 + half * 8 + v;
        if (l16 == 0 && gr < N) preds[gr] = p;
    }
}

// ---------------------------------------------------------------------------
// Loss: deterministic two-pass reduction of mean((preds - Y)^2).
// ---------------------------------------------------------------------------
#define LOSS_BLOCKS 256
__global__ void flk_loss_partial(const float* __restrict__ preds,
                                 const float* __restrict__ Y,
                                 float* __restrict__ partials, int N) {
    const int tid = threadIdx.x;
    float s = 0.f;
    for (int i = blockIdx.x * blockDim.x + tid; i < N;
         i += gridDim.x * blockDim.x) {
        const float d = preds[i] - Y[i];
        s += d * d;
    }
    s += __shfl_xor(s, 16, 32);
    s += __shfl_xor(s,  8, 32);
    s += __shfl_xor(s,  4, 32);
    s += __shfl_xor(s,  2, 32);
    s += __shfl_xor(s,  1, 32);
    __shared__ float red[8];
    if ((tid & 31) == 0) red[tid >> 5] = s;
    __syncthreads();
    if (tid == 0) {
        float tot = 0.f;
        #pragma unroll
        for (int w = 0; w < 8; ++w) tot += red[w];
        partials[blockIdx.x] = tot;
    }
}

__global__ void flk_loss_final(const float* __restrict__ partials,
                               float* __restrict__ out, int nb, float invN) {
    const int tid = threadIdx.x;
    float s = (tid < nb) ? partials[tid] : 0.f;
    s += __shfl_xor(s, 16, 32);
    s += __shfl_xor(s,  8, 32);
    s += __shfl_xor(s,  4, 32);
    s += __shfl_xor(s,  2, 32);
    s += __shfl_xor(s,  1, 32);
    __shared__ float red[8];
    if ((tid & 31) == 0) red[tid >> 5] = s;
    __syncthreads();
    if (tid == 0) {
        float tot = 0.f;
        #pragma unroll
        for (int w = 0; w < 8; ++w) tot += red[w];
        out[0] = tot * invN;
    }
}

// ---------------------------------------------------------------------------
// Host entry point.
// inputs: 0:X [N*D] 1:Y [N] 2:sigma [1] 3:penalty [1] 4:centers [M*D] 5:alpha [M]
// output: [loss, preds[N]] (float32)
// ---------------------------------------------------------------------------
extern "C" void kernel_launch(void* const* d_in, const int* in_sizes, int n_in,
                              void* d_out, int out_size, void* d_ws, size_t ws_size,
                              hipStream_t stream) {
    const float* X       = (const float*)d_in[0];
    const float* Y       = (const float*)d_in[1];
    const float* sigma_p = (const float*)d_in[2];
    const float* centers = (const float*)d_in[4];
    const float* alpha   = (const float*)d_in[5];

    const int N = in_sizes[1];               // Y is [N,1]
    const int M = in_sizes[5];               // alpha is [M,1]

    float* out_loss  = (float*)d_out;
    float* out_preds = out_loss + 1;

    // workspace layout
    char* w = (char*)d_ws;
    __bf16* Chi = (__bf16*)w;
    __bf16* Clo = Chi + (size_t)M * DCONST;
    float*  c2  = (float*)(Clo + (size_t)M * DCONST);
    float*  partials = c2 + M;

    // 1) prep centers (bf16 hi/lo split + |c|^2)
    flk_prep_centers<<<M, DCONST, 0, stream>>>(centers, Chi, Clo, c2);

    // 2) fused RBF mat-vec (async-DMA double-buffered B, register A)
    const int nblocks = (N + ROWS_PER_BLK - 1) / ROWS_PER_BLK;
    flk_rbf_mv<<<nblocks, MAIN_THREADS, 0, stream>>>(
        X, sigma_p, Chi, Clo, c2, alpha, out_preds, N, M);

    // 3) loss = mean((preds - Y)^2), deterministic two-pass reduce
    flk_loss_partial<<<LOSS_BLOCKS, 256, 0, stream>>>(out_preds, Y, partials, N);
    flk_loss_final<<<1, 256, 0, stream>>>(partials, out_loss, LOSS_BLOCKS,
                                          1.0f / (float)N);
}
